// ChebGNN_model_3169685864613
// MI455X (gfx1250) — matrix operand order repaired and checked by reading, not compiled, not run on hardware
//
#include <hip/hip_runtime.h>
#include <hip/hip_bf16.h>

// ---------------- types for WMMA fragments ----------------
typedef __attribute__((ext_vector_type(16))) __bf16 v16bf;
typedef __attribute__((ext_vector_type(8)))  __bf16 v8bf;
typedef __attribute__((ext_vector_type(4)))  __bf16 v4bf;
typedef __attribute__((ext_vector_type(8)))  float  v8f;

// Model dims (match reference)
#define NN      200000
#define EE      6400000
#define F_IN    300
#define KP1     320     // F_IN padded to multiple of 32 (max K over both layers)
#define H1      256
#define H2      128
#define HD      64
#define NLBL    20

// ---------------------------------------------------------------------------
// Convert x (N x 300 fp32) -> Xbf (N x 320 bf16, zero padded K)
// ---------------------------------------------------------------------------
__global__ void conv_x_bf16(const float* __restrict__ x, __bf16* __restrict__ xb) {
    size_t idx = (size_t)blockIdx.x * blockDim.x + threadIdx.x;
    size_t total = (size_t)NN * KP1;
    if (idx >= total) return;
    int r = (int)(idx / KP1);
    int c = (int)(idx % KP1);
    float v = (c < F_IN) ? x[(size_t)r * F_IN + c] : 0.f;
    xb[idx] = (__bf16)v;
}

// ---------------------------------------------------------------------------
// Convert W (2 x K x C fp32, row major) -> WT (2 x C x Kp bf16), zero padded K.
// WT layout is B^T: contiguous-K per output column (fragment friendly).
// ---------------------------------------------------------------------------
__global__ void conv_w_bf16(const float* __restrict__ W, __bf16* __restrict__ WT,
                            int K, int Kp, int Cn) {
    size_t idx = (size_t)blockIdx.x * blockDim.x + threadIdx.x;
    size_t total = (size_t)2 * Cn * Kp;
    if (idx >= total) return;
    int m  = (int)(idx / ((size_t)Cn * Kp));
    size_t rem = idx % ((size_t)Cn * Kp);
    int c  = (int)(rem / Kp);
    int kk = (int)(rem % Kp);
    float v = (kk < K) ? W[(size_t)m * K * Cn + (size_t)kk * Cn + c] : 0.f;
    WT[idx] = (__bf16)v;
}

__global__ void zero_f32(float* __restrict__ p, int n) {
    int i = blockIdx.x * blockDim.x + threadIdx.x;
    if (i < n) p[i] = 0.f;
}

// ---------------------------------------------------------------------------
// WMMA GEMM: C = A(MxKp, bf16) @ B(KpxCn)  with B given transposed (Cn x Kp).
// Block = 256 threads = 8 waves -> 128 rows x 64 cols per block.
// B panel (64 x Kp bf16, contiguous in BT) staged once per block into LDS via
// async global->LDS DMA (ASYNCcnt), then read as ds_load_b128 fragments.
// A-fragments register double-buffered so global loads overlap the WMMA pipe.
// outF (fp32) and outB (bf16) both get result +bias (if bias non-null). Either may be null.
// Fragment layouts per CDNA5 ISA 7.12.2 (wave32):
//   A 16x32 bf16 : lane L -> row L%16, halves[0..7]=K[(L>>4)*8 .. +7],
//                  halves[8..15]=K[16+(L>>4)*8 .. +7]
//   B 32x16 bf16 : lane L -> col L%16, halves in same K order as A
//   C 16x16 f32  : lane L -> col L%16, vgpr v -> row (L>>4)*8 + v
// ---------------------------------------------------------------------------
__global__ __launch_bounds__(256) void wmma_gemm_bf16(
    const __bf16* __restrict__ A, const __bf16* __restrict__ BT,
    const float* __restrict__ bias,
    float* __restrict__ outF, __bf16* __restrict__ outB,
    int M, int Kp, int Cn)
{
    __shared__ __align__(16) __bf16 sB[64 * KP1];   // 40 KB max panel

    const int lane = threadIdx.x & 31;
    const int wave = threadIdx.x >> 5;
    const int rowBase = blockIdx.x * 128 + wave * 16;
    const int colBase = blockIdx.y * 64;
    const int r  = lane & 15;
    const int hi = lane >> 4;            // 0 or 1

    // ---- stage B panel into LDS with async global->LDS copies ----
    {
        const char* gsrc   = (const char*)(BT + (size_t)colBase * Kp);
        unsigned    ldsOff = (unsigned)(uintptr_t)(&sB[0]); // low 32b = LDS addr
        int nChunks = (Kp * 64 * 2) / 16;                   // 16B chunks
        for (int c = threadIdx.x; c < nChunks; c += 256) {
            unsigned           lo = ldsOff + c * 16;
            unsigned long long ga = (unsigned long long)(uintptr_t)(gsrc + c * 16);
            asm volatile("global_load_async_to_lds_b128 %0, %1, off"
                         :: "v"(lo), "v"(ga) : "memory");
        }
        asm volatile("s_wait_asynccnt 0x0" ::: "memory");
        __syncthreads();
    }

    int arow = rowBase + r;
    if (arow >= M) arow = M - 1;         // clamp: EXEC must stay all-ones for WMMA
    const __bf16* aPtr = A + (size_t)arow * Kp + hi * 8;

    v8f acc[4] = {};

    // prologue A-fragment
    v8bf alo = *(const v8bf*)(aPtr + 0);
    v8bf ahi = *(const v8bf*)(aPtr + 16);
    v16bf aF = __builtin_shufflevector(alo, ahi,
        0,1,2,3,4,5,6,7,8,9,10,11,12,13,14,15);

    for (int kc = 0; kc < Kp; kc += 32) {
        v16bf aN = aF;
        if (kc + 32 < Kp) {              // prefetch next A fragment (overlaps WMMA)
            v8bf nlo = *(const v8bf*)(aPtr + kc + 32);
            v8bf nhi = *(const v8bf*)(aPtr + kc + 48);
            aN = __builtin_shufflevector(nlo, nhi,
                0,1,2,3,4,5,6,7,8,9,10,11,12,13,14,15);
        }
#pragma unroll
        for (int j = 0; j < 4; ++j) {
            const __bf16* bp = &sB[(size_t)(16 * j + r) * Kp + kc + hi * 8];
            v8bf blo = *(const v8bf*)(bp);       // ds_load_b128
            v8bf bhi = *(const v8bf*)(bp + 16);  // ds_load_b128
            v16bf bfrag = __builtin_shufflevector(blo, bhi,
                0,1,2,3,4,5,6,7,8,9,10,11,12,13,14,15);
            acc[j] = __builtin_amdgcn_wmma_f32_16x16x32_bf16(
                false, aF, false, bfrag, (short)0, acc[j], false, false);
        }
        aF = aN;
    }

#pragma unroll
    for (int j = 0; j < 4; ++j) {
        int col = colBase + 16 * j + r;
        float bv = bias ? bias[col] : 0.f;
#pragma unroll
        for (int v = 0; v < 8; ++v) {
            int row = rowBase + hi * 8 + v;
            if (row < M) {
                float val = acc[j][v] + bv;
                if (outF) outF[(size_t)row * Cn + col] = val;
                if (outB) outB[(size_t)row * Cn + col] = (__bf16)val;
            }
        }
    }
}

// ---------------------------------------------------------------------------
// Edge scatter: for each edge e, T[dst[e]][:] += w[e] * P[src[e]][:]
// P is bf16 (halved gather bandwidth), accumulation fp32 atomics.
// C/4 threads per edge, 4 features per thread (8-byte bf16 gather).
// ---------------------------------------------------------------------------
template <int C>
__global__ __launch_bounds__(256) void scatter_edges(
    const int* __restrict__ src, const int* __restrict__ dst,
    const float* __restrict__ w, const __bf16* __restrict__ P,
    float* __restrict__ T, int nEdges)
{
    constexpr int LPE = C / 4;          // lanes per edge
    constexpr int EPB = 256 / LPE;      // edges per block
    int e  = blockIdx.x * EPB + (int)(threadIdx.x / LPE);
    if (e >= nEdges) return;
    int lp = (int)(threadIdx.x % LPE);

    int   s  = src[e];
    int   d  = dst[e];
    float ww = w[e];

    v4bf pv = *(const v4bf*)(P + (size_t)s * C + lp * 4);
    float* t = T + (size_t)d * C + lp * 4;
#pragma unroll
    for (int i = 0; i < 4; ++i)
        atomicAdd(t + i, ww * (float)pv[i]);
}

// ---------------------------------------------------------------------------
// h_bf16 = bf16(relu(T))   (feeds next layer's WMMA A-matrix)
// ---------------------------------------------------------------------------
__global__ void relu_to_bf16(const float* __restrict__ T, __bf16* __restrict__ hb,
                             size_t total) {
    size_t idx = (size_t)blockIdx.x * blockDim.x + threadIdx.x;
    if (idx >= total) return;
    hb[idx] = (__bf16)fmaxf(T[idx], 0.f);
}

// ---------------------------------------------------------------------------
// g[col] += sum over 64-row chunk of relu(T2[row][col]);  128 cols.
// ---------------------------------------------------------------------------
__global__ __launch_bounds__(128) void relu_pool(const float* __restrict__ T,
                                                 float* __restrict__ g, int M) {
    int col = threadIdx.x;              // 0..127
    int rowBase = blockIdx.x * 64;
    float s = 0.f;
    for (int rr = 0; rr < 64; ++rr) {
        int row = rowBase + rr;
        if (row < M) s += fmaxf(T[(size_t)row * H2 + col], 0.f);
    }
    atomicAdd(&g[col], s);
}

// ---------------------------------------------------------------------------
// MLP head + softmax (single block of 64 threads; tiny).
// ---------------------------------------------------------------------------
__global__ __launch_bounds__(64) void mlp_head(
    const float* __restrict__ g,
    const float* __restrict__ Wd1, const float* __restrict__ bd1,
    const float* __restrict__ Wd2, const float* __restrict__ bd2,
    float* __restrict__ out)
{
    __shared__ float gh[HD];
    __shared__ float z[NLBL];
    int t = threadIdx.x;

    float s = bd1[t];
    for (int k = 0; k < H2; ++k) s += g[k] * Wd1[(size_t)k * HD + t];
    gh[t] = fmaxf(s, 0.f);
    __syncthreads();

    if (t < NLBL) {
        float zz = bd2[t];
        for (int k = 0; k < HD; ++k) zz += gh[k] * Wd2[(size_t)k * NLBL + t];
        z[t] = zz;
    }
    __syncthreads();

    if (t == 0) {
        float m = z[0];
        for (int i = 1; i < NLBL; ++i) m = fmaxf(m, z[i]);
        float sum = 0.f;
        float e[NLBL];
        for (int i = 0; i < NLBL; ++i) { e[i] = __expf(z[i] - m); sum += e[i]; }
        float inv = 1.f / sum;
        for (int i = 0; i < NLBL; ++i) out[i] = e[i] * inv;
    }
}

// ---------------------------------------------------------------------------
extern "C" void kernel_launch(void* const* d_in, const int* in_sizes, int n_in,
                              void* d_out, int out_size, void* d_ws, size_t ws_size,
                              hipStream_t stream) {
    (void)in_sizes; (void)n_in; (void)out_size; (void)ws_size;

    const float* x   = (const float*)d_in[0];
    const int*   esr = (const int*)  d_in[1];
    const int*   eds = (const int*)  d_in[2];
    const float* ew  = (const float*)d_in[3];
    const float* W1  = (const float*)d_in[4];
    const float* b1  = (const float*)d_in[5];
    const float* W2  = (const float*)d_in[6];
    const float* b2  = (const float*)d_in[7];
    const float* Wd1 = (const float*)d_in[8];
    const float* bd1 = (const float*)d_in[9];
    const float* Wd2 = (const float*)d_in[10];
    const float* bd2 = (const float*)d_in[11];
    float* out = (float*)d_out;

    char* ws = (char*)d_ws;
    // Phase-aliased workspace layout
    size_t oXBF = 0;                                   // N*320 bf16 (128.0 MB); reused as h1bf (N*256 bf16)
    size_t oT   = oXBF + (size_t)NN * KP1 * 2;         // T1: N*256 f32 (204.8 MB); reused as T2 (N*128 f32)
    size_t oP   = oT   + (size_t)NN * H1 * 4;          // P1: N*256 bf16 (102.4 MB); reused as P2 (N*128 bf16)
    size_t oW1T = oP   + (size_t)NN * H1 * 2;          // 2*256*320 bf16
    size_t oW2T = oW1T + (size_t)2 * H1 * KP1 * 2;     // 2*128*256 bf16
    size_t oG   = oW2T + (size_t)2 * H2 * H1 * 2;      // 128 f32

    __bf16* Xbf  = (__bf16*)(ws + oXBF);
    __bf16* h1bf = (__bf16*)(ws + oXBF);               // alias (after Xbf dead)
    float*  T1   = (float*) (ws + oT);
    float*  T2   = (float*) (ws + oT);                 // alias (after T1 dead)
    __bf16* P1   = (__bf16*)(ws + oP);
    __bf16* P2   = (__bf16*)(ws + oP);                 // alias (after P1 dead)
    __bf16* W1T  = (__bf16*)(ws + oW1T);
    __bf16* W2T  = (__bf16*)(ws + oW2T);
    float*  g    = (float*) (ws + oG);

    // ---- precision conversion / packing ----
    {
        size_t tot = (size_t)NN * KP1;
        conv_x_bf16<<<(unsigned)((tot + 255) / 256), 256, 0, stream>>>(x, Xbf);
        size_t tw1 = (size_t)2 * H1 * KP1;
        conv_w_bf16<<<(unsigned)((tw1 + 255) / 256), 256, 0, stream>>>(W1, W1T, F_IN, KP1, H1);
        size_t tw2 = (size_t)2 * H2 * H1;
        conv_w_bf16<<<(unsigned)((tw2 + 255) / 256), 256, 0, stream>>>(W2, W2T, H1, H1, H2);
        zero_f32<<<1, 128, 0, stream>>>(g, H2);
    }

    dim3 blk(256);

    // ---- Layer 1 ----
    {
        dim3 grid((NN + 127) / 128, H1 / 64);
        // T1 = X @ W1[0] + b1
        wmma_gemm_bf16<<<grid, blk, 0, stream>>>(Xbf, W1T, b1, T1, nullptr, NN, KP1, H1);
        // P1 = bf16(X @ W1[1])   (propagate AFTER matmul: 256 dims instead of 300)
        wmma_gemm_bf16<<<grid, blk, 0, stream>>>(Xbf, W1T + (size_t)H1 * KP1, nullptr,
                                                 nullptr, P1, NN, KP1, H1);
        // T1[dst] += w * P1[src]
        int epb = 256 / (H1 / 4);
        scatter_edges<H1><<<(EE + epb - 1) / epb, blk, 0, stream>>>(esr, eds, ew, P1, T1, EE);
        // h1 = bf16(relu(T1))
        size_t tot = (size_t)NN * H1;
        relu_to_bf16<<<(unsigned)((tot + 255) / 256), 256, 0, stream>>>(T1, h1bf, tot);
    }

    // ---- Layer 2 ----
    {
        dim3 grid((NN + 127) / 128, H2 / 64);
        wmma_gemm_bf16<<<grid, blk, 0, stream>>>(h1bf, W2T, b2, T2, nullptr, NN, H1, H2);
        wmma_gemm_bf16<<<grid, blk, 0, stream>>>(h1bf, W2T + (size_t)H2 * H1, nullptr,
                                                 nullptr, P2, NN, H1, H2);
        int epb = 256 / (H2 / 4);
        scatter_edges<H2><<<(EE + epb - 1) / epb, blk, 0, stream>>>(esr, eds, ew, P2, T2, EE);
        // pool: g = sum over rows of relu(T2)
        relu_pool<<<(NN + 63) / 64, 128, 0, stream>>>(T2, g, NN);
    }

    // ---- Head ----
    mlp_head<<<1, HD, 0, stream>>>(g, Wd1, bd1, Wd2, bd2, out);
}
